// GCNNet_51307679318513
// MI455X (gfx1250) — matrix operand order repaired and checked by reading, not compiled, not run on hardware
//
#include <hip/hip_runtime.h>
#include <math.h>

typedef __attribute__((ext_vector_type(16))) _Float16 v16h;
typedef __attribute__((ext_vector_type(8)))  float    v8f;

#define N_GRAPHS  25
#define NODES_PER 2000
#define NN        (N_GRAPHS * NODES_PER)   // 50000 nodes
#define EE        800000                   // edges
#define DD        128                      // hidden dim
#define NLAYERS   4
#define WP_PER_LAYER (8*4*32*8)            // u32 words of packed W per layer

// ---------------------------------------------------------------------------
// h[row] = emb[labels[row]] + sinusoidal PE (scattered through perms)
// thread = (graph-local position j, frequency pair d2)
// ---------------------------------------------------------------------------
__global__ void k_init_h(const int* __restrict__ labels,
                         const int* __restrict__ perms,
                         const float* __restrict__ emb,
                         float* __restrict__ h)
{
    int tid = blockIdx.x * blockDim.x + threadIdx.x;   // NN*64 threads
    if (tid >= NN * 64) return;
    int i  = tid >> 6;          // flat (g, j)
    int d2 = tid & 63;          // frequency pair index
    int g  = i / NODES_PER;
    int j  = i - g * NODES_PER;
    int row = g * NODES_PER + perms[i];
    int lab = labels[row];
    // div = exp(2*d2 * (-ln(10000)/D)); position starts at 1
    float div = expf((float)(2 * d2) * (-9.210340371976184f / (float)DD));
    float ang = (float)(j + 1) * div;
    h[(long)row * DD + 2 * d2]     = emb[lab * DD + 2 * d2]     + sinf(ang);
    h[(long)row * DD + 2 * d2 + 1] = emb[lab * DD + 2 * d2 + 1] + cosf(ang);
}

// ---------------------------------------------------------------------------
// degree accumulation + inverse-sqrt transform
// ---------------------------------------------------------------------------
__global__ void k_deg(const int* __restrict__ src, const int* __restrict__ dst,
                      float* __restrict__ deg_out, float* __restrict__ deg_in)
{
    int e = blockIdx.x * blockDim.x + threadIdx.x;
    if (e >= EE) return;
    atomicAdd(&deg_out[src[e]], 1.0f);
    atomicAdd(&deg_in[dst[e]],  1.0f);
}

__global__ void k_isq(float* __restrict__ a, float* __restrict__ b)
{
    int i = blockIdx.x * blockDim.x + threadIdx.x;
    if (i >= NN) return;
    a[i] = rsqrtf(fmaxf(a[i], 1.0f));
    b[i] = rsqrtf(fmaxf(b[i], 1.0f));
}

// ---------------------------------------------------------------------------
// Pack Ws (f32 [L,128,128]) into a per-lane f16 B-matrix VGPR layout (contiguous k, same as A)
// for v_wmma_f32_16x16x32_f16, so the GEMM streams it with b128 loads.
// B 32x16 (KxN), k order used here: VGPR v, lanes 0-15: N=lane,    K=2v,2v+1
//                         VGPR v, lanes 16-31: N=lane-16, K=16+2v,+1
// index = ((((l*8 + t)*4 + c)*32 + L)*8 + v)
// ---------------------------------------------------------------------------
__global__ void k_wprep(const float* __restrict__ Ws, unsigned int* __restrict__ wp)
{
    int tid = blockIdx.x * blockDim.x + threadIdx.x;
    if (tid >= NLAYERS * WP_PER_LAYER) return;
    int v = tid & 7;
    int L = (tid >> 3) & 31;
    int c = (tid >> 8) & 3;
    int t = (tid >> 10) & 7;
    int l = tid >> 13;
    int hi = (L >= 16) ? 1 : 0;
    int n  = t * 16 + (L & 15);
    int k0 = c * 32 + (hi ? 16 : 0) + 2 * v;
    union { _Float16 h[2]; unsigned int u; } pk;
    pk.h[0] = (_Float16)Ws[((long)l * DD + k0)     * DD + n];
    pk.h[1] = (_Float16)Ws[((long)l * DD + k0 + 1) * DD + n];
    wp[tid] = pk.u;
}

// ---------------------------------------------------------------------------
// Edge gather/scatter: m[dst] += h[src] * isq_src[src]
// One wave per edge, one float4 per lane (32*4 = 128). L2-resident h.
// ---------------------------------------------------------------------------
__global__ void k_scatter(const int* __restrict__ src, const int* __restrict__ dst,
                          const float* __restrict__ h, const float* __restrict__ isq_src,
                          float* __restrict__ m)
{
    int wave = (blockIdx.x * blockDim.x + threadIdx.x) >> 5;
    int lane = threadIdx.x & 31;
    if (wave >= EE) return;
    int s = src[wave];
    int d = dst[wave];
    float w = isq_src[s];
    float4 v = ((const float4*)(h + (long)s * DD))[lane];
    float* mp = m + (long)d * DD + lane * 4;
    atomicAdd(mp + 0, v.x * w);
    atomicAdd(mp + 1, v.y * w);
    atomicAdd(mp + 2, v.z * w);
    atomicAdd(mp + 3, v.w * w);
}

// ---------------------------------------------------------------------------
// GCN layer GEMM: h += relu((m * isq_dst) @ W + b), via WMMA f16->f32.
// One wave per 16-row tile; A held in regs (4 K-chunks), 8 N-tiles of B
// streamed from packed layout; 32 v_wmma per wave.
// A 16x32 16-bit layout: lanes 0-15 row=lane, VGPR v: K = 2v,2v+1 (same k order as packed B);
// lanes 16-31 row=lane-16, same +16.
// C/D layout: lane L holds N=L&15; VGPR j holds M = j + (L>=16 ? 8 : 0).
// ---------------------------------------------------------------------------
__global__ void k_gcn_gemm(const float* __restrict__ m, const float* __restrict__ isq_dst,
                           const unsigned int* __restrict__ wp, const float* __restrict__ bias,
                           float* __restrict__ h)
{
    const int ntiles = NN / 16;   // 3125
    int wave = (blockIdx.x * blockDim.x + threadIdx.x) >> 5;
    int lane = threadIdx.x & 31;
    if (wave >= ntiles) return;
    int hi = lane >> 4;
    int r  = wave * 16 + (lane & 15);
    float scale = isq_dst[r];
    const float* mrow = m + (long)r * DD;

    v16h a[4];
#pragma unroll
    for (int c = 0; c < 4; c++) {
#pragma unroll
        for (int v = 0; v < 8; v++) {
            int k0 = c * 32 + (hi ? 16 : 0) + 2 * v;
            a[c][2 * v]     = (_Float16)(mrow[k0]     * scale);
            a[c][2 * v + 1] = (_Float16)(mrow[k0 + 1] * scale);
        }
    }

#pragma unroll
    for (int t = 0; t < 8; t++) {
        v8f acc = {};
#pragma unroll
        for (int c = 0; c < 4; c++) {
            v16h b = *(const v16h*)(wp + (((long)(t * 4 + c) * 32 + lane) * 8));
            acc = __builtin_amdgcn_wmma_f32_16x16x32_f16(
                false, a[c], false, b, (short)0, acc, false, false);
        }
        int n = t * 16 + (lane & 15);
        float bn = bias[n];
#pragma unroll
        for (int j = 0; j < 8; j++) {
            int row = wave * 16 + j + (hi ? 8 : 0);
            float val = fmaxf(acc[j] + bn, 0.0f);
            h[(long)row * DD + n] += val;   // residual
        }
    }
}

// ---------------------------------------------------------------------------
// MLP readout: 128 -> 64 -> 32 -> 128, one block (128 threads) per row.
// ---------------------------------------------------------------------------
__global__ void k_readout(const float* __restrict__ h,
                          const float* __restrict__ w1, const float* __restrict__ b1,
                          const float* __restrict__ w2, const float* __restrict__ b2,
                          const float* __restrict__ w3, const float* __restrict__ b3,
                          float* __restrict__ out)
{
    __shared__ float hs[DD];
    __shared__ float x1[64];
    __shared__ float x2[32];
    int row = blockIdx.x;
    int j = threadIdx.x;
    hs[j] = h[(long)row * DD + j];
    __syncthreads();
    if (j < 64) {
        float acc = b1[j];
        for (int k = 0; k < DD; k++) acc += hs[k] * w1[k * 64 + j];
        x1[j] = fmaxf(acc, 0.0f);
    }
    __syncthreads();
    if (j < 32) {
        float acc = b2[j];
        for (int k = 0; k < 64; k++) acc += x1[k] * w2[k * 32 + j];
        x2[j] = fmaxf(acc, 0.0f);
    }
    __syncthreads();
    float acc = b3[j];
    for (int k = 0; k < 32; k++) acc += x2[k] * w3[k * DD + j];
    out[(long)row * DD + j] = acc;
}

// ---------------------------------------------------------------------------
extern "C" void kernel_launch(void* const* d_in, const int* in_sizes, int n_in,
                              void* d_out, int out_size, void* d_ws, size_t ws_size,
                              hipStream_t stream)
{
    (void)in_sizes; (void)n_in; (void)out_size; (void)ws_size;
    const int*   labels = (const int*)d_in[0];
    const int*   src    = (const int*)d_in[1];
    const int*   dst    = (const int*)d_in[2];
    const int*   perms  = (const int*)d_in[3];
    const float* emb    = (const float*)d_in[4];
    const float* Ws     = (const float*)d_in[5];
    const float* bs     = (const float*)d_in[6];
    const float* w1     = (const float*)d_in[7];
    const float* b1     = (const float*)d_in[8];
    const float* w2     = (const float*)d_in[9];
    const float* b2     = (const float*)d_in[10];
    const float* w3     = (const float*)d_in[11];
    const float* b3     = (const float*)d_in[12];
    float* out = (float*)d_out;

    // workspace layout
    float* h     = (float*)d_ws;                 // NN*DD f32
    float* isq_s = h + (size_t)NN * DD;          // NN f32
    float* isq_d = isq_s + NN;                   // NN f32
    unsigned int* wp = (unsigned int*)(isq_d + NN); // NLAYERS*WP_PER_LAYER u32
    float* msg = out;                            // reuse d_out as message buffer

    hipMemsetAsync(isq_s, 0, (size_t)2 * NN * sizeof(float), stream);
    k_init_h<<<(NN * 64 + 255) / 256, 256, 0, stream>>>(labels, perms, emb, h);
    k_deg<<<(EE + 255) / 256, 256, 0, stream>>>(src, dst, isq_s, isq_d);
    k_isq<<<(NN + 255) / 256, 256, 0, stream>>>(isq_s, isq_d);
    k_wprep<<<(NLAYERS * WP_PER_LAYER + 255) / 256, 256, 0, stream>>>(Ws, wp);

    for (int l = 0; l < NLAYERS; l++) {
        hipMemsetAsync(msg, 0, (size_t)NN * DD * sizeof(float), stream);
        k_scatter<<<((size_t)EE * 32 + 255) / 256, 256, 0, stream>>>(
            src, dst, h, isq_s, msg);
        k_gcn_gemm<<<((NN / 16) + 7) / 8, 256, 0, stream>>>(
            msg, isq_d, wp + (size_t)l * WP_PER_LAYER, bs + (size_t)l * DD, h);
    }

    k_readout<<<NN, DD, 0, stream>>>(h, w1, b1, w2, b2, w3, b3, out);
}
